// Decoder_25615184954195
// MI455X (gfx1250) — matrix operand order repaired and checked by reading, not compiled, not run on hardware
//
#include <hip/hip_runtime.h>

// Problem constants (match reference)
#define LSTM_HID    1024
#define LSTM_GATES  4096   // 4*HID
#define LSTM_BATCH  256
#define LSTM_LAYERS 2
#define LSTM_STEPS  32     // 29 + 3
#define HEAD_OUT    512

typedef __attribute__((ext_vector_type(8)))  __bf16 v8bf;
typedef __attribute__((ext_vector_type(16))) __bf16 v16bf;
typedef __attribute__((ext_vector_type(8)))  float  v8f;

// ---------------------------------------------------------------------------
// WMMA helpers (CDNA5 wave32 layouts, ISA 7.12.2)
// ---------------------------------------------------------------------------

// A-matrix 16x32 bf16 tile. Row M = m (same for both lane halves).
// lane<16:  elems 0..15 = K k0+0..15   (klo=0)
// lane>=16: elems 0..15 = K k0+16..31  (klo=16)
__device__ __forceinline__ v16bf load_a16x32(const __bf16* __restrict__ base,
                                             int m, int k0, int klo) {
  const __bf16* p = base + (size_t)m * LSTM_HID + k0 + klo;
  v8bf lo = *(const v8bf*)p;
  v8bf hi = *(const v8bf*)(p + 8);
  return __builtin_shufflevector(lo, hi,
                                 0, 1, 2, 3, 4, 5, 6, 7,
                                 8, 9, 10, 11, 12, 13, 14, 15);
}

// B-matrix 32x16 bf16 tile from row-major W[n,k] (W^T is free):
// lane holds column n = n0+(lane&15); elems 0..15 = K k0+khalf..+15
__device__ __forceinline__ v16bf load_b32x16(const __bf16* __restrict__ W,
                                             int n, int k0, int khalf) {
  return *(const v16bf*)(W + (size_t)n * LSTM_HID + k0 + khalf);
}

__device__ __forceinline__ v8f wmma_bf16(v16bf a, v16bf b, v8f c) {
  return __builtin_amdgcn_wmma_f32_16x16x32_bf16(
      /*neg_a=*/false, a, /*neg_b=*/false, b,
      /*c_mod=*/(short)0, c, /*reuse_a=*/false, /*reuse_b=*/false);
}

// ---------------------------------------------------------------------------
// Fused dual GEMM: gates[B,4H] = X[B,H] @ Wih^T + H[B,H] @ Whh^T
//
// Grid: 64 blocks (one per 64-wide N strip), 256 threads (8 waves).
// Each WG covers all M=256 rows; wave w owns rows 32w..32w+31 (2 M-tiles)
// and the whole 64-wide N strip (4 subtiles) -> 8 f32 accumulators.
// B strips (Wih+Whh, 8KB per 32-K chunk) live in double-buffered LDS shared
// by all 8 waves (weights leave L2 exactly once per GEMM). The k-loop is
// manually unrolled 2x so each half-body statically addresses one LDS
// buffer (no runtime cur/nxt index -> no accumulator register rotation),
// while the next chunk's B strip is prefetched into registers.
//
// LDS layout per buffer: 2 matrices x 64 rows x 40 halves (80B padded rows);
// padding makes the 16-lane column reads walk distinct bank groups.
// ---------------------------------------------------------------------------
#define LDSB_ROW   40                    // halves per row (32 data + 8 pad)
#define LDSB_MAT   (64 * LDSB_ROW)       // 2560 halves per matrix block
#define LDSB_BUF   (2 * LDSB_MAT)        // 5120 halves per buffer

__global__ __launch_bounds__(256)
void lstm_gates_gemm(const __bf16* __restrict__ X, const __bf16* __restrict__ H,
                     const __bf16* __restrict__ Wih, const __bf16* __restrict__ Whh,
                     float* __restrict__ gates) {
  __shared__ __bf16 lds[2][LDSB_BUF];

  const int t     = threadIdx.x;
  const int lane  = t & 31;
  const int wave  = t >> 5;
  const int n0    = blockIdx.x << 6;          // 64-wide N strip
  const int m0    = wave << 5;                // rows 32w .. 32w+31
  const int ma    = m0 + (lane & 15);
  const int klo   = (lane & 16) ? 8 : 0;
  const int khalf = (lane & 16) ? 16 : 0;
  const int lcol  = lane & 15;

  // staging assignment: 256 threads move 8KB (2 matrices x 64 rows x 64B)
  const int sm    = t >> 7;                   // 0 = Wih, 1 = Whh
  const int srow  = (t & 127) >> 1;           // 0..63
  const int spart = t & 1;                    // which 32B half of the row
  const __bf16* sW = sm ? Whh : Wih;
  const __bf16* sgp = sW + (size_t)(n0 + srow) * LSTM_HID + spart * 16;
  const int sdst = sm * LDSB_MAT + srow * LDSB_ROW + spart * 16;

  v8f a00 = {}, a01 = {}, a02 = {}, a03 = {};
  v8f a10 = {}, a11 = {}, a12 = {}, a13 = {};

  // one chunk of the K loop: 4 A tiles from global, 8 B tiles from one
  // statically-addressed LDS buffer, 16 WMMAs accumulating in place.
  auto mma_chunk = [&](const __bf16* buf, int k0) {
    v16bf ax0 = load_a16x32(X, ma,      k0, khalf);
    v16bf ah0 = load_a16x32(H, ma,      k0, khalf);
    v16bf ax1 = load_a16x32(X, ma + 16, k0, khalf);
    v16bf ah1 = load_a16x32(H, ma + 16, k0, khalf);
#pragma unroll
    for (int s = 0; s < 4; ++s) {
      const int r = (s << 4) + lcol;
      v16bf bih = *(const v16bf*)&buf[0 * LDSB_MAT + r * LDSB_ROW + khalf];
      v16bf bhh = *(const v16bf*)&buf[1 * LDSB_MAT + r * LDSB_ROW + khalf];
      switch (s) {
        case 0: a00 = wmma_bf16(ax0, bih, a00); a00 = wmma_bf16(ah0, bhh, a00);
                a10 = wmma_bf16(ax1, bih, a10); a10 = wmma_bf16(ah1, bhh, a10);
                break;
        case 1: a01 = wmma_bf16(ax0, bih, a01); a01 = wmma_bf16(ah0, bhh, a01);
                a11 = wmma_bf16(ax1, bih, a11); a11 = wmma_bf16(ah1, bhh, a11);
                break;
        case 2: a02 = wmma_bf16(ax0, bih, a02); a02 = wmma_bf16(ah0, bhh, a02);
                a12 = wmma_bf16(ax1, bih, a12); a12 = wmma_bf16(ah1, bhh, a12);
                break;
        default:a03 = wmma_bf16(ax0, bih, a03); a03 = wmma_bf16(ah0, bhh, a03);
                a13 = wmma_bf16(ax1, bih, a13); a13 = wmma_bf16(ah1, bhh, a13);
                break;
      }
    }
  };

  // prologue: stage chunk 0 into buffer 0
  *(v16bf*)&lds[0][sdst] = *(const v16bf*)(sgp);
  __syncthreads();

#pragma unroll 1
  for (int kc2 = 0; kc2 < 16; ++kc2) {
    const int k0 = kc2 << 6;  // even chunk base (64 K per outer iteration)

    // ---- half A: consume lds[0] (chunk k0), stage chunk k0+32 -> lds[1]
    v16bf stg1 = *(const v16bf*)(sgp + (size_t)(k0 + 32));
    mma_chunk(&lds[0][0], k0);
    *(v16bf*)&lds[1][sdst] = stg1;
    __syncthreads();

    // ---- half B: consume lds[1] (chunk k0+32), stage chunk k0+64 -> lds[0]
    v16bf stg0;
    if (kc2 < 15) stg0 = *(const v16bf*)(sgp + (size_t)(k0 + 64));
    mma_chunk(&lds[1][0], k0 + 32);
    if (kc2 < 15) *(v16bf*)&lds[0][sdst] = stg0;
    __syncthreads();
  }

  // epilogue: C/D layout -> gates
  const int rlo = (lane >> 4) << 3;  // +8 rows for upper lane half
  {
    float* gp = gates + (size_t)(m0 + rlo) * LSTM_GATES + n0 + lcol;
#pragma unroll
    for (int r = 0; r < 8; ++r) {
      float* row = gp + (size_t)r * LSTM_GATES;
      row[0]  = a00[r];
      row[16] = a01[r];
      row[32] = a02[r];
      row[48] = a03[r];
    }
  }
  {
    float* gp = gates + (size_t)(m0 + 16 + rlo) * LSTM_GATES + n0 + lcol;
#pragma unroll
    for (int r = 0; r < 8; ++r) {
      float* row = gp + (size_t)r * LSTM_GATES;
      row[0]  = a10[r];
      row[16] = a11[r];
      row[32] = a12[r];
      row[48] = a13[r];
    }
  }
}

// ---------------------------------------------------------------------------
// Head GEMM: out[B,512] = X[B,H] @ Wlin^T + b_lin  (tiny: 0.1% of FLOPs)
// Grid: 16 blocks (M tiles); 8 waves x 64N = 512 N per block.
// ---------------------------------------------------------------------------
__global__ __launch_bounds__(256)
void head_gemm(const __bf16* __restrict__ X, const __bf16* __restrict__ W,
               const float* __restrict__ bias, float* __restrict__ out) {
  const int lane  = threadIdx.x & 31;
  const int wave  = threadIdx.x >> 5;
  const int m0    = blockIdx.x << 4;
  const int n0    = wave << 6;
  const int ma    = m0 + (lane & 15);
  const int klo   = (lane & 16) ? 8 : 0;
  const int khalf = (lane & 16) ? 16 : 0;
  const int ncol  = n0 + (lane & 15);

  v8f acc0 = {}, acc1 = {}, acc2 = {}, acc3 = {};
  for (int k0 = 0; k0 < LSTM_HID; k0 += 32) {
    v16bf ax = load_a16x32(X, ma, k0, khalf);
    acc0 = wmma_bf16(ax, load_b32x16(W, ncol,      k0, khalf), acc0);
    acc1 = wmma_bf16(ax, load_b32x16(W, ncol + 16, k0, khalf), acc1);
    acc2 = wmma_bf16(ax, load_b32x16(W, ncol + 32, k0, khalf), acc2);
    acc3 = wmma_bf16(ax, load_b32x16(W, ncol + 48, k0, khalf), acc3);
  }

  const int rbase = m0 + ((lane >> 4) << 3);
  const float b0 = bias[ncol], b1 = bias[ncol + 16];
  const float b2 = bias[ncol + 32], b3 = bias[ncol + 48];
  float* op = out + (size_t)rbase * HEAD_OUT + ncol;
#pragma unroll
  for (int r = 0; r < 8; ++r) {
    float* row = op + (size_t)r * HEAD_OUT;
    row[0]  = acc0[r] + b0;
    row[16] = acc1[r] + b1;
    row[32] = acc2[r] + b2;
    row[48] = acc3[r] + b3;
  }
}

// ---------------------------------------------------------------------------
// Pointwise LSTM cell update. Gate order i,f,g,o (PyTorch).
// ---------------------------------------------------------------------------
__device__ __forceinline__ float sigmoidf_(float x) {
  return 1.0f / (1.0f + __expf(-x));
}

__global__ __launch_bounds__(256)
void lstm_pointwise(const float* __restrict__ gates,
                    const float* __restrict__ bih, const float* __restrict__ bhh,
                    float* __restrict__ c, __bf16* __restrict__ h) {
  const int idx = blockIdx.x * blockDim.x + threadIdx.x;
  if (idx >= LSTM_BATCH * LSTM_HID) return;
  const int b = idx >> 10;
  const int j = idx & (LSTM_HID - 1);
  const float* g = gates + (size_t)b * LSTM_GATES;

  const float ig = g[j]                + bih[j]                + bhh[j];
  const float fg = g[j + LSTM_HID]     + bih[j + LSTM_HID]     + bhh[j + LSTM_HID];
  const float gg = g[j + 2 * LSTM_HID] + bih[j + 2 * LSTM_HID] + bhh[j + 2 * LSTM_HID];
  const float og = g[j + 3 * LSTM_HID] + bih[j + 3 * LSTM_HID] + bhh[j + 3 * LSTM_HID];

  const float cn = sigmoidf_(fg) * c[idx] + sigmoidf_(ig) * tanhf(gg);
  c[idx] = cn;
  h[idx] = (__bf16)(sigmoidf_(og) * tanhf(cn));
}

// ---------------------------------------------------------------------------
// f32 -> bf16 conversion (grid-stride)
// ---------------------------------------------------------------------------
__global__ void f32_to_bf16_kernel(const float* __restrict__ s,
                                   __bf16* __restrict__ d, int n) {
  for (int i = blockIdx.x * blockDim.x + threadIdx.x; i < n;
       i += gridDim.x * blockDim.x) {
    d[i] = (__bf16)s[i];
  }
}

// ---------------------------------------------------------------------------
// Host driver
// ---------------------------------------------------------------------------
extern "C" void kernel_launch(void* const* d_in, const int* in_sizes, int n_in,
                              void* d_out, int out_size, void* d_ws, size_t ws_size,
                              hipStream_t stream) {
  const float* in_x    = (const float*)d_in[0];  // [256,1024]
  const float* in_h    = (const float*)d_in[1];  // [2,256,1024]
  const float* in_c    = (const float*)d_in[2];  // [2,256,1024]
  const float* in_Wih  = (const float*)d_in[3];  // [2,4096,1024]
  const float* in_Whh  = (const float*)d_in[4];  // [2,4096,1024]
  const float* in_bih  = (const float*)d_in[5];  // [2,4096]
  const float* in_bhh  = (const float*)d_in[6];  // [2,4096]
  const float* in_Wlin = (const float*)d_in[7];  // [512,1024]
  const float* in_blin = (const float*)d_in[8];  // [512]
  float* out = (float*)d_out;                    // [256,512]

  const size_t W_ELEMS    = (size_t)LSTM_LAYERS * LSTM_GATES * LSTM_HID;
  const size_t WLIN_ELEMS = (size_t)HEAD_OUT * LSTM_HID;
  const size_t BH_ELEMS   = (size_t)LSTM_BATCH * LSTM_HID;
  const size_t ST_ELEMS   = (size_t)LSTM_LAYERS * BH_ELEMS;
  const size_t G_ELEMS    = (size_t)LSTM_BATCH * LSTM_GATES;

  char* ws = (char*)d_ws;
  size_t off = 0;
  auto alloc = [&](size_t bytes) -> char* {
    char* p = ws + off;
    off += (bytes + 255) & ~(size_t)255;
    return p;
  };
  __bf16* wih_bf  = (__bf16*)alloc(W_ELEMS * 2);
  __bf16* whh_bf  = (__bf16*)alloc(W_ELEMS * 2);
  __bf16* wlin_bf = (__bf16*)alloc(WLIN_ELEMS * 2);
  __bf16* x_bf    = (__bf16*)alloc(BH_ELEMS * 2);
  __bf16* h_bf    = (__bf16*)alloc(ST_ELEMS * 2);
  float*  c_ws    = (float*)alloc(ST_ELEMS * 4);
  float*  gates   = (float*)alloc(G_ELEMS * 4);
  if (off > ws_size) return;

  // one-time (per call) conversions / state init, all on `stream`
  f32_to_bf16_kernel<<<4096, 256, 0, stream>>>(in_Wih,  wih_bf,  (int)W_ELEMS);
  f32_to_bf16_kernel<<<4096, 256, 0, stream>>>(in_Whh,  whh_bf,  (int)W_ELEMS);
  f32_to_bf16_kernel<<<1024, 256, 0, stream>>>(in_Wlin, wlin_bf, (int)WLIN_ELEMS);
  f32_to_bf16_kernel<<<1024, 256, 0, stream>>>(in_x,    x_bf,    (int)BH_ELEMS);
  f32_to_bf16_kernel<<<1024, 256, 0, stream>>>(in_h,    h_bf,    (int)ST_ELEMS);
  hipMemcpyAsync(c_ws, in_c, ST_ELEMS * sizeof(float),
                 hipMemcpyDeviceToDevice, stream);

  const int gemmBlocks = LSTM_GATES / 64;        // 64 N strips
  const int pwBlocks   = (int)(BH_ELEMS / 256);  // 1024

  __bf16* h0 = h_bf;
  __bf16* h1 = h_bf + BH_ELEMS;
  float*  c0 = c_ws;
  float*  c1 = c_ws + BH_ELEMS;
  const size_t WL = W_ELEMS / LSTM_LAYERS;       // per-layer weight elems

  for (int step = 0; step < LSTM_STEPS; ++step) {
    const __bf16* x_in = (step == 0) ? x_bf : h1;  // feedback: x_{t+1} = h1_t

    // layer 0
    lstm_gates_gemm<<<gemmBlocks, 256, 0, stream>>>(x_in, h0, wih_bf, whh_bf,
                                                    gates);
    lstm_pointwise<<<pwBlocks, 256, 0, stream>>>(gates, in_bih, in_bhh, c0, h0);

    // layer 1 (input = new h0)
    lstm_gates_gemm<<<gemmBlocks, 256, 0, stream>>>(h0, h1, wih_bf + WL,
                                                    whh_bf + WL, gates);
    lstm_pointwise<<<pwBlocks, 256, 0, stream>>>(gates, in_bih + LSTM_GATES,
                                                 in_bhh + LSTM_GATES, c1, h1);
  }

  // linear head on final stack output
  head_gemm<<<LSTM_BATCH / 16, 256, 0, stream>>>(h1, wlin_bf, in_blin, out);
}
